// MultiDomainPLEFENDModel_30193620091263
// MI455X (gfx1250) — compile-verified, matrix operand order checked by reading, not run on hardware
//
#include <hip/hip_runtime.h>
#include <hip/hip_bf16.h>

typedef __bf16 bf16_t;
typedef __attribute__((ext_vector_type(16))) __bf16 v16bf;
typedef __attribute__((ext_vector_type(8)))  __bf16 v8bf;
typedef __attribute__((ext_vector_type(8)))  float  v8f;

#define NBATCH  32
#define SEQ     197
#define PADL    224
#define DMODEL  768
#define NEXP    60
#define FKCH    64
#define FEATN   320
#define MLPH    384
#define GATEE   18
#define NEPD    6

// ---- workspace layout ----
// xbf : [2][32][224][768] bf16                = 11,010,048 halves
// wp  : [2][60][2016 frags][32 lanes][16]     = 123,863,040 halves
// feat: [2][60][32][320] f32                  = 1,228,800 floats
#define XBF_ELEMS 11010048ull
#define WP_ELEMS  123863040ull

// CDNA5 async global->LDS staging (ASYNCcnt), guarded so the file always builds.
#if defined(__HIP_DEVICE_COMPILE__) && \
    __has_builtin(__builtin_amdgcn_global_load_async_to_lds_b128) && \
    __has_builtin(__builtin_amdgcn_s_wait_asynccnt)
#define USE_ASYNC_LDS 1
typedef int v4i_t __attribute__((vector_size(16)));
typedef __attribute__((address_space(1))) v4i_t* g_v4i_p;   // global int4*
typedef __attribute__((address_space(3))) v4i_t* l_v4i_p;   // LDS int4*
#else
#define USE_ASYNC_LDS 0
#endif

// ---------------------------------------------------------------------------
// Kernel 1: f32 -> bf16 copy of features, padded to 224 rows (zeros past 197)
// ---------------------------------------------------------------------------
__global__ __launch_bounds__(256)
void convert_x(const float* __restrict__ xt, const float* __restrict__ xi,
               bf16_t* __restrict__ xbf)
{
    size_t idx = (size_t)blockIdx.x * 256 + threadIdx.x;   // exactly XBF_ELEMS
    int    dcol = (int)(idx % DMODEL);
    size_t r    = idx / DMODEL;
    int    row  = (int)(r % PADL);
    size_t mb   = r / PADL;            // 0..63
    int    b    = (int)(mb % NBATCH);
    int    m    = (int)(mb / NBATCH);
    const float* src = m ? xi : xt;
    float v = (row < SEQ) ? src[((size_t)b * SEQ + row) * DMODEL + dcol] : 0.f;
    xbf[idx] = (bf16_t)v;
}

// ---------------------------------------------------------------------------
// Kernel 2: pre-shuffle conv weights into per-lane WMMA B-fragment layout.
// ---------------------------------------------------------------------------
struct WArgs { const float* w[10]; };   // [m*5 + ki]

__global__ __launch_bounds__(256)
void prepack_w(WArgs args, bf16_t* __restrict__ wp)
{
    const int ktab[5] = {1, 2, 3, 5, 10};
    const int preF[6] = {0, 96, 288, 576, 1056, 2016};

    size_t idx  = (size_t)blockIdx.x * 256 + threadIdx.x;  // exactly 7,741,440
    int    lane = (int)(idx & 31);
    size_t f    = idx >> 5;
    int    fi   = (int)(f % 2016);
    int    ee   = (int)(f / 2016);
    int    e    = ee % NEXP;
    int    m    = ee / NEXP;

    int ki = 0;
    while (fi >= preF[ki + 1]) ki++;
    int within = fi - preF[ki];
    int kstep  = within >> 2;
    int nt     = within & 3;
    int k      = ktab[ki];
    int n      = nt * 16 + (lane & 15);
    int g      = lane >> 4;

    const float* wsrc = args.w[m * 5 + ki];
    v16bf out;
#pragma unroll
    for (int h = 0; h < 16; ++h) {
        int off = h + 8 * g + ((h < 8) ? 0 : 8);
        int K   = kstep * 32 + off;
        int j   = K / DMODEL;
        int d   = K - j * DMODEL;
        out[h]  = (bf16_t)wsrc[((size_t)(e * FKCH + n) * DMODEL + d) * k + j];
    }
    *(v16bf*)(wp + idx * 16) = out;
}

// ---------------------------------------------------------------------------
// Kernel 3: WMMA implicit-GEMM conv + fused global max-pool + bias.
// Grid: 2*60*5*2 = 1200 WGs of 256 threads (8 wave32s). Wave w owns batches
// b0 = bh*16+w and b1 = b0+8 (2 M-tiles -> 8 WMMAs per staged B tile).
// B tile (32x64 bf16 = 4KB) is double-buffered in LDS; with CDNA5 async
// global->LDS loads the staging needs no VGPR round-trip and one barrier/step.
// ---------------------------------------------------------------------------
__global__ __launch_bounds__(256)
void conv_moe_wmma(const bf16_t* __restrict__ xbf, const bf16_t* __restrict__ wp,
                   const float* __restrict__ cb_text, const float* __restrict__ cb_image,
                   float* __restrict__ feat)
{
    __shared__ __align__(16) unsigned char ldsB[2][4096];
    const int ktab[5] = {1, 2, 3, 5, 10};
    const int preF[5] = {0, 96, 288, 576, 1056};

    int wi = blockIdx.x;
    int bh = wi & 1;  wi >>= 1;
    int ki = wi % 5;  wi /= 5;
    int e  = wi % NEXP;
    int m  = wi / NEXP;

    const int k   = ktab[ki];
    const int P   = SEQ - k + 1;          // valid output positions
    const int nPT = (P + 15) >> 4;        // 16-position tiles
    const int nK  = 24 * k;               // K-steps of 32 over 768*k (even)

    const int tid  = threadIdx.x;
    const int lane = tid & 31;
    const int w    = tid >> 5;
    const int g    = lane >> 4;
    const int lm   = lane & 15;
    const int b0   = bh * 16 + w;
    const int b1   = b0 + 8;

    const unsigned char* wpk =
        (const unsigned char*)(wp + ((size_t)(m * NEXP + e) * 2016 + preF[ki]) * 512);
    const float* cb = (m == 0) ? cb_text : cb_image;

    float rmax[2][4] = {{-3.0e38f, -3.0e38f, -3.0e38f, -3.0e38f},
                        {-3.0e38f, -3.0e38f, -3.0e38f, -3.0e38f}};

    for (int pt = 0; pt < nPT; ++pt) {
        const int p0 = pt * 16;
        const bf16_t* arow0 = xbf + ((size_t)(m * NBATCH + b0) * PADL + p0 + lm) * DMODEL + g * 8;
        const bf16_t* arow1 = xbf + ((size_t)(m * NBATCH + b1) * PADL + p0 + lm) * DMODEL + g * 8;

        v8f acc[2][4] = {};

#if USE_ASYNC_LDS
        // Prologue: async-stage k-step 0 into buffer 0.
        __builtin_amdgcn_global_load_async_to_lds_b128(
            (g_v4i_p)(wpk + (size_t)tid * 16),
            (l_v4i_p)(&ldsB[0][tid * 16]), 0, 0);
        __builtin_amdgcn_s_wait_asynccnt(0);
        __syncthreads();
#else
        uint4 pre = *(const uint4*)(wpk + (size_t)tid * 16);
#endif

        for (int ks = 0; ks < nK; ++ks) {
            const int buf = ks & 1;
#if USE_ASYNC_LDS
            if (ks + 1 < nK)   // async-prefetch next B tile into the other buffer
                __builtin_amdgcn_global_load_async_to_lds_b128(
                    (g_v4i_p)(wpk + (size_t)(ks + 1) * 4096 + (size_t)tid * 16),
                    (l_v4i_p)(&ldsB[buf ^ 1][tid * 16]), 0, 0);
#else
            *(uint4*)(&ldsB[buf][tid * 16]) = pre;
            __syncthreads();
            if (ks + 1 < nK)
                pre = *(const uint4*)(wpk + (size_t)(ks + 1) * 4096 + (size_t)tid * 16);
#endif
            // Load all 4 B fragments into distinct registers (one wait, then
            // 8 back-to-back WMMAs instead of load->wait->wmma serialization).
            v16bf Bv[4];
#pragma unroll
            for (int nt = 0; nt < 4; ++nt) {
                union { v16bf v; v8bf h[2]; } Bf;
                const unsigned char* bsrc = &ldsB[buf][nt * 1024 + lane * 32];
                Bf.h[0] = *(const v8bf*)(bsrc);
                Bf.h[1] = *(const v8bf*)(bsrc + 16);
                Bv[nt]  = Bf.v;
            }
            union { v16bf v; v8bf h[2]; } A0, A1;
            A0.h[0] = *(const v8bf*)(arow0 + ks * 32);
            A0.h[1] = *(const v8bf*)(arow0 + ks * 32 + 16);
            A1.h[0] = *(const v8bf*)(arow1 + ks * 32);
            A1.h[1] = *(const v8bf*)(arow1 + ks * 32 + 16);

#pragma unroll
            for (int nt = 0; nt < 4; ++nt)
                acc[0][nt] = __builtin_amdgcn_wmma_f32_16x16x32_bf16(
                    false, A0.v, false, Bv[nt], (short)0, acc[0][nt], false, false);
#pragma unroll
            for (int nt = 0; nt < 4; ++nt)
                acc[1][nt] = __builtin_amdgcn_wmma_f32_16x16x32_bf16(
                    false, A1.v, false, Bv[nt], (short)0, acc[1][nt], false, false);

#if USE_ASYNC_LDS
            __builtin_amdgcn_s_wait_asynccnt(0);
            __syncthreads();
#else
            __syncthreads();
#endif
        }

        // fused max-pool: rows M = r + 8*g (C/D layout), mask tail positions
#pragma unroll
        for (int mi = 0; mi < 2; ++mi) {
#pragma unroll
            for (int nt = 0; nt < 4; ++nt) {
                float local = -3.0e38f;
#pragma unroll
                for (int r = 0; r < 8; ++r) {
                    int pos = p0 + r + 8 * g;
                    float v = acc[mi][nt][r];
                    local = (pos < P) ? fmaxf(local, v) : local;
                }
                local         = fmaxf(local, __shfl_xor(local, 16, 32));
                rmax[mi][nt]  = fmaxf(rmax[mi][nt], local);
            }
        }
    }

    if (lane < 16) {
#pragma unroll
        for (int mi = 0; mi < 2; ++mi) {
            int b = mi ? b1 : b0;
#pragma unroll
            for (int nt = 0; nt < 4; ++nt) {
                int fk = nt * 16 + lane;
                feat[((size_t)(m * NEXP + e) * NBATCH + b) * FEATN + ki * FKCH + fk] =
                    rmax[mi][nt] + cb[(ki * NEXP + e) * FKCH + fk];
            }
        }
    }
}

// ---------------------------------------------------------------------------
// Kernel 4: attention pool + per-domain gate + soft-MoE combine + head.
// ---------------------------------------------------------------------------
struct TailArgs {
    const float* x[2];
    const float* masks;
    const int*   category;
    const float* aw[2];  const float* ab[2];
    const float* gw1[2]; const float* gb1[2];
    const float* gw2[2]; const float* gb2[2];
    const float* cw1[2]; const float* cb1[2];
    const float* cw2[2]; const float* cb2[2];
    const float* demb;
    const float* feat;
    float*       out;
};

__global__ __launch_bounds__(256)
void tail_kernel(TailArgs a)
{
    __shared__ float s[SEQ];
    __shared__ float p[SEQ];
    __shared__ float gin[2 * DMODEL];
    __shared__ float hbuf[DMODEL];
    __shared__ float gate[GATEE];
    __shared__ float comb[FEATN];
    __shared__ float hh[MLPH];
    __shared__ float red[2];

    int m   = blockIdx.x >> 5;
    int b   = blockIdx.x & 31;
    int tid = threadIdx.x;
    int d   = a.category[b];
    const float* x  = a.x[m];
    const float* aw = a.aw[m];
    float        ab = a.ab[m][0];

    for (int l = tid; l < SEQ; l += 256) {
        float acc = ab;
        const float* xr = x + ((size_t)b * SEQ + l) * DMODEL;
        for (int c = 0; c < DMODEL; ++c) acc += xr[c] * aw[c];
        if (m == 0 && a.masks[b * SEQ + l] <= 0.f) acc = -1e9f;
        s[l] = acc;
    }
    __syncthreads();
    if (tid == 0) {
        float mx = -3e38f;
        for (int l = 0; l < SEQ; ++l) mx = fmaxf(mx, s[l]);
        float ss = 0.f;
        for (int l = 0; l < SEQ; ++l) { float ev = __expf(s[l] - mx); p[l] = ev; ss += ev; }
        red[0] = 1.f / ss;
    }
    __syncthreads();
    float inv = red[0];
    for (int c = tid; c < DMODEL; c += 256) {
        float acc = 0.f;
        for (int l = 0; l < SEQ; ++l) acc += p[l] * x[((size_t)b * SEQ + l) * DMODEL + c];
        gin[c]          = acc * inv;
        gin[DMODEL + c] = a.demb[(size_t)d * DMODEL + c];
    }
    __syncthreads();
    for (int o = tid; o < DMODEL; o += 256) {
        float acc = a.gb1[m][(size_t)d * DMODEL + o];
        const float* wcol = a.gw1[m] + (size_t)d * 2 * DMODEL * DMODEL + o;
        for (int i = 0; i < 2 * DMODEL; ++i) acc += gin[i] * wcol[(size_t)i * DMODEL];
        hbuf[o] = acc / (1.f + __expf(-acc));
    }
    __syncthreads();
    if (tid < GATEE) {
        float acc = a.gb2[m][d * GATEE + tid];
        const float* w2 = a.gw2[m] + (size_t)d * DMODEL * GATEE + tid;
        for (int o = 0; o < DMODEL; ++o) acc += hbuf[o] * w2[(size_t)o * GATEE];
        gate[tid] = acc;
    }
    __syncthreads();
    if (tid == 0) {
        float mx = gate[0];
        for (int j = 1; j < GATEE; ++j) mx = fmaxf(mx, gate[j]);
        float ss = 0.f;
        for (int j = 0; j < GATEE; ++j) { float ev = __expf(gate[j] - mx); gate[j] = ev; ss += ev; }
        red[1] = 1.f / ss;
    }
    __syncthreads();
    float ginv = red[1];
    for (int fi = tid; fi < FEATN; fi += 256) {
        float acc = 0.f;
        for (int j = 0; j < GATEE; ++j) {
            int ge = (j < NEPD) ? (d * NEPD + j) : (48 + j - NEPD);
            acc += gate[j] * ginv *
                   a.feat[((size_t)(m * NEXP + ge) * NBATCH + b) * FEATN + fi];
        }
        comb[fi] = acc;
    }
    __syncthreads();
    for (int o = tid; o < MLPH; o += 256) {
        float acc = a.cb1[m][(size_t)d * MLPH + o];
        const float* w1 = a.cw1[m] + (size_t)d * FEATN * MLPH + o;
        for (int fi = 0; fi < FEATN; ++fi) acc += comb[fi] * w1[(size_t)fi * MLPH];
        hh[o] = fmaxf(acc, 0.f);
    }
    __syncthreads();
    if (tid == 0) {
        float acc = a.cb2[m][d];
        const float* w2 = a.cw2[m] + (size_t)d * MLPH;
        for (int o = 0; o < MLPH; ++o) acc += hh[o] * w2[o];
        a.out[m * NBATCH + b] = 1.f / (1.f + __expf(-acc));
    }
}

// ---------------------------------------------------------------------------
extern "C" void kernel_launch(void* const* d_in, const int* in_sizes, int n_in,
                              void* d_out, int out_size, void* d_ws, size_t ws_size,
                              hipStream_t stream)
{
    (void)in_sizes; (void)n_in; (void)out_size; (void)ws_size;

    const float* text_x  = (const float*)d_in[0];
    const float* image_x = (const float*)d_in[1];
    const float* masks   = (const float*)d_in[2];
    const int*   categ   = (const int*)d_in[3];

    bf16_t* xbf  = (bf16_t*)d_ws;
    bf16_t* wp   = xbf + XBF_ELEMS;
    float*  feat = (float*)(wp + WP_ELEMS);

    convert_x<<<43008, 256, 0, stream>>>(text_x, image_x, xbf);

    WArgs wa;
    for (int m = 0; m < 2; ++m)
        for (int ki = 0; ki < 5; ++ki)
            wa.w[m * 5 + ki] = (const float*)d_in[4 + m * 16 + ki];
    prepack_w<<<30240, 256, 0, stream>>>(wa, wp);

    conv_moe_wmma<<<1200, 256, 0, stream>>>(xbf, wp,
        (const float*)d_in[9], (const float*)d_in[25], feat);

    TailArgs ta;
    ta.x[0] = text_x; ta.x[1] = image_x; ta.masks = masks; ta.category = categ;
    for (int m = 0; m < 2; ++m) {
        int base = 4 + m * 16;
        ta.gw1[m] = (const float*)d_in[base + 6];
        ta.gb1[m] = (const float*)d_in[base + 7];
        ta.gw2[m] = (const float*)d_in[base + 8];
        ta.gb2[m] = (const float*)d_in[base + 9];
        ta.cw1[m] = (const float*)d_in[base + 10];
        ta.cb1[m] = (const float*)d_in[base + 11];
        ta.cw2[m] = (const float*)d_in[base + 12];
        ta.cb2[m] = (const float*)d_in[base + 13];
        ta.aw[m]  = (const float*)d_in[base + 14];
        ta.ab[m]  = (const float*)d_in[base + 15];
    }
    ta.demb = (const float*)d_in[36];
    ta.feat = feat;
    ta.out  = (float*)d_out;
    tail_kernel<<<64, 256, 0, stream>>>(ta);
}
